// CrossAttention_44538810859971
// MI455X (gfx1250) — compile-verified, matrix operand order checked by reading, not run on hardware
//
#include <hip/hip_runtime.h>
#include <hip/hip_bf16.h>

typedef __attribute__((ext_vector_type(16))) _Float16     v16h;
typedef __attribute__((ext_vector_type(8)))  _Float16     v8h;
typedef __attribute__((ext_vector_type(8)))  float        v8f;
typedef __attribute__((ext_vector_type(4)))  float        vf4;
typedef __attribute__((ext_vector_type(4)))  unsigned int u32x4;
typedef __attribute__((ext_vector_type(8)))  int          i32x8;
typedef __attribute__((ext_vector_type(4)))  int          i32x4;

#define HW   25600
#define Hdim 160
#define Wdim 160
#define CQ   192
#define NB   8
#define NH   6
#define CPH  32

#if __has_builtin(__builtin_amdgcn_tensor_load_to_lds) && __has_builtin(__builtin_amdgcn_s_wait_tensorcnt)
#define USE_TDM 1
#else
#define USE_TDM 0
#endif

__device__ inline v8f v8zero() {
    v8f z = {0.f,0.f,0.f,0.f,0.f,0.f,0.f,0.f};
    return z;
}

__device__ inline v8f wmma_f16(v16h a, v16h b, v8f c) {
    return __builtin_amdgcn_wmma_f32_16x16x32_f16(false, a, false, b, (short)0, c, false, false);
}

// A-style fragment from a row-major f32 matrix row (contiguous K), with on-the-fly cvt.
//   elems 0..7  -> k = 8*hsel + e ;  elems 8..15 -> k = 16 + 8*hsel + (e-8)
__device__ inline v16h load_a16(const float* __restrict__ row, int hsel) {
    const vf4* r1 = (const vf4*)(row + 8 * hsel);
    const vf4* r2 = (const vf4*)(row + 16 + 8 * hsel);
    vf4 a0 = r1[0], a1 = r1[1], a2 = r2[0], a3 = r2[1];
    v16h f;
    f[0]=(_Float16)a0[0]; f[1]=(_Float16)a0[1]; f[2]=(_Float16)a0[2]; f[3]=(_Float16)a0[3];
    f[4]=(_Float16)a1[0]; f[5]=(_Float16)a1[1]; f[6]=(_Float16)a1[2]; f[7]=(_Float16)a1[3];
    f[8]=(_Float16)a2[0]; f[9]=(_Float16)a2[1]; f[10]=(_Float16)a2[2]; f[11]=(_Float16)a2[3];
    f[12]=(_Float16)a3[0]; f[13]=(_Float16)a3[1]; f[14]=(_Float16)a3[2]; f[15]=(_Float16)a3[3];
    return f;
}

// Same fragment shape, but from a pre-converted f16 matrix: two b128 loads, no cvt.
__device__ inline v16h load_a16h(const _Float16* __restrict__ row, int hsel) {
    const v8h* r1 = (const v8h*)(row + 8 * hsel);
    const v8h* r2 = (const v8h*)(row + 16 + 8 * hsel);
    v8h a = r1[0], b = r2[0];
    v16h f;
#pragma unroll
    for (int e = 0; e < 8; ++e) { f[e] = a[e]; f[8 + e] = b[e]; }
    return f;
}

// B fragment = A-style layout of B^T gathered from a column of a row-major f32 matrix.
__device__ inline v16h load_bT_strided(const float* __restrict__ col, size_t stride, int hsel) {
    v16h f;
#pragma unroll
    for (int e = 0; e < 16; ++e) {
        int k = (e < 8 ? e : e + 8) + 8 * hsel;
        f[e] = (_Float16)col[(size_t)k * stride];
    }
    return f;
}

// B fragment from an LDS-staged [32 x 16] f32 tile (row-major k*16+n).
__device__ inline v16h load_bT_lds(const float* sb, int nl, int hsel) {
    v16h f;
#pragma unroll
    for (int e = 0; e < 16; ++e) {
        int k = (e < 8 ? e : e + 8) + 8 * hsel;
        f[e] = (_Float16)sb[k * 16 + nl];
    }
    return f;
}

#if USE_TDM
// Issue a TDM load of a [tile_dim1=32 rows x tile_dim0=16 cols] f32 tile,
// source stride HW elements, into LDS at lds_byte. D# packing per ISA 8.3-8.5.
__device__ inline void tdm_load_tile(const float* gaddr, unsigned lds_byte) {
    unsigned long long ga = (unsigned long long)(uintptr_t)gaddr;
    u32x4 g0;
    g0[0] = 1u;                                            // count=1, no gather
    g0[1] = lds_byte;                                      // lds_addr
    g0[2] = (unsigned)ga;                                  // global_addr[31:0]
    g0[3] = (unsigned)((ga >> 32) & 0x01FFFFFFu) | (2u << 30);  // addr[56:32] | type=2
    i32x8 g1;
    g1[0] = (int)(2u << 16);          // workgroup_mask=0, data_size=2 (4B)
    g1[1] = (int)((unsigned)HW << 16);// tensor_dim0[15:0] in [31:16]
    g1[2] = (int)(192u << 16);        // tensor_dim0 hi=0 | tensor_dim1=192
    g1[3] = (int)(16u << 16);         // tensor_dim1 hi=0 | tile_dim0=16
    g1[4] = 32;                       // tile_dim1=32, tile_dim2=0
    g1[5] = HW;                       // tensor_dim0_stride[31:0]
    g1[6] = 0;                        // stride0 hi, stride1 lo
    g1[7] = 0;
    i32x4 gz = {0, 0, 0, 0};
#if defined(__clang_major__) && (__clang_major__ >= 23)
    i32x8 gz8 = {0, 0, 0, 0, 0, 0, 0, 0};
    __builtin_amdgcn_tensor_load_to_lds(g0, g1, gz, gz, gz8, 0);
#else
    __builtin_amdgcn_tensor_load_to_lds(g0, g1, gz, gz, 0);
#endif
}
#endif

// ---------------------------------------------------------------------------
// 1x1 conv as GEMM: Y[b] = W(Cout x 192) * X[b](192 x HW).
// f16 WMMA, f32 accum; W pre-converted to f16; X tiles staged to LDS by TDM
// (double-buffered per wave) when available.
// block = 128 (4 waves); each wave: one 16-wide N tile, full M strip.
// ---------------------------------------------------------------------------
template<int MTILES>
__global__ __launch_bounds__(128) void pw_gemm(const _Float16* __restrict__ Wh,
                                               const float* __restrict__ X,
                                               float* __restrict__ Y) {
    constexpr int Cout = MTILES * 16;
    const int lane = threadIdx.x & 31;
    const int wave = threadIdx.x >> 5;
    const int b    = blockIdx.y;
    const int m_   = lane & 15;
    const int hsel = lane >> 4;
    const int n0   = (blockIdx.x * 4 + wave) * 16;
    const int n    = n0 + m_;

    const float* __restrict__ Xb = X + (size_t)b * CQ * HW;
    float* __restrict__       Yb = Y + (size_t)b * Cout * HW;

    v8f acc[MTILES];
#pragma unroll
    for (int i = 0; i < MTILES; ++i) acc[i] = v8zero();

#if USE_TDM
    __shared__ float stage[4][2][32 * 16];   // [wave][buf][k*16+n] = 16 KB
    const unsigned lds_base =
        (unsigned)(uintptr_t)(&stage[wave][0][0]);
    // prologue: start K-step 0
    tdm_load_tile(Xb + (size_t)0 * HW + n0, lds_base);
#pragma unroll
    for (int kk = 0; kk < 6; ++kk) {           // K = 192 = 6 x 32
        if (kk < 5) {
            tdm_load_tile(Xb + (size_t)(kk + 1) * 32 * HW + n0,
                          lds_base + (unsigned)(((kk + 1) & 1) * 32 * 16 * 4));
            __builtin_amdgcn_s_wait_tensorcnt(1);
        } else {
            __builtin_amdgcn_s_wait_tensorcnt(0);
        }
        const float* sb = &stage[wave][kk & 1][0];
        v16h bf = load_bT_lds(sb, m_, hsel);
        const int k0 = kk * 32;
#pragma unroll
        for (int mt = 0; mt < MTILES; ++mt) {
            v16h af = load_a16h(Wh + (size_t)(mt * 16 + m_) * CQ + k0, hsel);
            acc[mt] = wmma_f16(af, bf, acc[mt]);
        }
    }
#else
    for (int kk = 0; kk < 6; ++kk) {
        const int k0 = kk * 32;
        v16h bf = load_bT_strided(Xb + (size_t)k0 * HW + n, HW, hsel);
#pragma unroll
        for (int mt = 0; mt < MTILES; ++mt) {
            v16h af = load_a16h(Wh + (size_t)(mt * 16 + m_) * CQ + k0, hsel);
            acc[mt] = wmma_f16(af, bf, acc[mt]);
        }
    }
#endif

#pragma unroll
    for (int mt = 0; mt < MTILES; ++mt) {
#pragma unroll
        for (int r = 0; r < 8; ++r) {
            const int m = mt * 16 + r + 8 * hsel;
            Yb[(size_t)m * HW + n] = acc[mt][r];
        }
    }
}

// ---------------------------------------------------------------------------
// f32 -> f16 conversion for small constant operands (weights).
// ---------------------------------------------------------------------------
__global__ __launch_bounds__(256) void cvt_f16(const float* __restrict__ src,
                                               _Float16* __restrict__ dst, int n) {
    int i = blockIdx.x * 256 + threadIdx.x;
    if (i < n) dst[i] = (_Float16)src[i];
}

// ---------------------------------------------------------------------------
// Depthwise 3x3, pad 1. One thread per output pixel.
// ---------------------------------------------------------------------------
__global__ __launch_bounds__(256) void dwconv3x3(const float* __restrict__ X,
                                                 const float* __restrict__ Wd,
                                                 float* __restrict__ Y, int C) {
    const size_t idx = (size_t)blockIdx.x * 256 + threadIdx.x;
    const int xw = (int)(idx % Wdim);
    const int yh = (int)((idx / Wdim) % Hdim);
    const size_t plane = idx / HW;             // b*C + c
    const int c = (int)(plane % C);
    const float* __restrict__ base = X + plane * HW;
    const float* __restrict__ wk   = Wd + (size_t)c * 9;
    float s = 0.f;
#pragma unroll
    for (int dy = -1; dy <= 1; ++dy) {
        const int yy = yh + dy;
        if (yy < 0 || yy >= Hdim) continue;
#pragma unroll
        for (int dx = -1; dx <= 1; ++dx) {
            const int xx = xw + dx;
            if (xx < 0 || xx >= Wdim) continue;
            s = fmaf(base[yy * Wdim + xx], wk[(dy + 1) * 3 + (dx + 1)], s);
        }
    }
    Y[idx] = s;
}

// ---------------------------------------------------------------------------
// Per-(b,c) plane inverse L2 norm. grid = NB*192; Cbuf = channels in buffer.
// ---------------------------------------------------------------------------
__global__ __launch_bounds__(256) void plane_invnorm(const float* __restrict__ X,
                                                     float* __restrict__ inv, int Cbuf) {
    const int p = blockIdx.x;
    const int b = p / CQ, c = p % CQ;
    const float* __restrict__ base = X + ((size_t)b * Cbuf + c) * HW;
    float s = 0.f;
    for (int i = threadIdx.x; i < HW; i += 256) {
        float v = base[i];
        s = fmaf(v, v, s);
    }
    __shared__ float red[256];
    red[threadIdx.x] = s;
    __syncthreads();
    for (int o = 128; o > 0; o >>= 1) {
        if ((int)threadIdx.x < o) red[threadIdx.x] += red[threadIdx.x + o];
        __syncthreads();
    }
    if (threadIdx.x == 0) inv[p] = 1.0f / fmaxf(sqrtf(red[0]), 1e-12f);
}

// ---------------------------------------------------------------------------
// S = q k^T over HW, scaled by invq*invk*temp, row softmax -> P (f16).
// grid = NB*NH blocks of 256 (8 waves); K split across waves, LDS reduce.
// ---------------------------------------------------------------------------
__global__ __launch_bounds__(256) void attn_scores(const float* __restrict__ Q,
                                                   const float* __restrict__ KV,
                                                   const float* __restrict__ invq,
                                                   const float* __restrict__ invk,
                                                   const float* __restrict__ temp,
                                                   _Float16* __restrict__ P) {
    const int bh = blockIdx.x;
    const int b = bh / NH, h = bh % NH;
    const float* __restrict__ Qb = Q  + ((size_t)b * CQ     + h * CPH) * HW;
    const float* __restrict__ Kb = KV + ((size_t)b * 2 * CQ + h * CPH) * HW;
    const int lane = threadIdx.x & 31;
    const int wave = threadIdx.x >> 5;
    const int m_   = lane & 15;
    const int hsel = lane >> 4;

    __shared__ float S[CPH * CPH];
    for (int i = threadIdx.x; i < CPH * CPH; i += 256) S[i] = 0.f;
    __syncthreads();

    v8f acc[2][2];
    acc[0][0] = v8zero(); acc[0][1] = v8zero();
    acc[1][0] = v8zero(); acc[1][1] = v8zero();

    for (int t = 0; t < 100; ++t) {          // 8 waves * 100 * 32 = 25600
        const int k0 = wave * 3200 + t * 32;
        v16h aq0 = load_a16(Qb + (size_t)m_        * HW + k0, hsel);
        v16h aq1 = load_a16(Qb + (size_t)(16 + m_) * HW + k0, hsel);
        v16h bk0 = load_a16(Kb + (size_t)m_        * HW + k0, hsel);
        v16h bk1 = load_a16(Kb + (size_t)(16 + m_) * HW + k0, hsel);
        acc[0][0] = wmma_f16(aq0, bk0, acc[0][0]);
        acc[0][1] = wmma_f16(aq0, bk1, acc[0][1]);
        acc[1][0] = wmma_f16(aq1, bk0, acc[1][0]);
        acc[1][1] = wmma_f16(aq1, bk1, acc[1][1]);
    }
#pragma unroll
    for (int i = 0; i < 2; ++i)
#pragma unroll
        for (int j = 0; j < 2; ++j)
#pragma unroll
            for (int r = 0; r < 8; ++r) {
                const int m = i * 16 + r + 8 * hsel;
                const int d = j * 16 + m_;
                atomicAdd(&S[m * CPH + d], acc[i][j][r]);
            }
    __syncthreads();

    if (threadIdx.x < CPH) {
        const int r = threadIdx.x;
        const float iq = invq[b * CQ + h * CPH + r] * temp[h];
        float row[CPH];
        float mx = -3.4e38f;
        for (int d = 0; d < CPH; ++d) {
            float v = S[r * CPH + d] * iq * invk[b * CQ + h * CPH + d];
            row[d] = v;
            mx = fmaxf(mx, v);
        }
        float sum = 0.f;
        for (int d = 0; d < CPH; ++d) {
            float e = __expf(row[d] - mx);
            row[d] = e;
            sum += e;
        }
        const float rs = 1.0f / sum;
        for (int d = 0; d < CPH; ++d)
            P[(size_t)bh * (CPH * CPH) + r * CPH + d] = (_Float16)(row[d] * rs);
    }
}

// ---------------------------------------------------------------------------
// out = P(32x32) @ V(32xHW). P pre-converted f16; K=32 -> one WMMA per tile.
// grid = (HW/16/32, NB*NH), block = 256; each wave handles 4 N tiles.
// ---------------------------------------------------------------------------
__global__ __launch_bounds__(256) void attn_apply(const _Float16* __restrict__ P,
                                                  const float* __restrict__ KV,
                                                  float* __restrict__ O) {
    const int bh = blockIdx.y;
    const int b = bh / NH, h = bh % NH;
    const _Float16* __restrict__ Pm = P + (size_t)bh * (CPH * CPH);
    const float* __restrict__ Vb = KV + ((size_t)b * 2 * CQ + CQ + h * CPH) * HW;
    float* __restrict__       Ob = O  + ((size_t)b * CQ          + h * CPH) * HW;
    const int lane = threadIdx.x & 31;
    const int wave = threadIdx.x >> 5;
    const int m_   = lane & 15;
    const int hsel = lane >> 4;

    v16h ap0 = load_a16h(Pm + (size_t)m_        * CPH, hsel);
    v16h ap1 = load_a16h(Pm + (size_t)(16 + m_) * CPH, hsel);

    for (int t = 0; t < 4; ++t) {
        const int n0 = ((blockIdx.x * 8 + wave) * 4 + t) * 16;
        const int n  = n0 + m_;
        v16h bv = load_bT_strided(Vb + n, HW, hsel);
        v8f o0 = wmma_f16(ap0, bv, v8zero());
        v8f o1 = wmma_f16(ap1, bv, v8zero());
#pragma unroll
        for (int r = 0; r < 8; ++r) {
            Ob[(size_t)(r + 8 * hsel) * HW + n]      = o0[r];
            Ob[(size_t)(16 + r + 8 * hsel) * HW + n] = o1[r];
        }
    }
}

// ---------------------------------------------------------------------------
extern "C" void kernel_launch(void* const* d_in, const int* in_sizes, int n_in,
                              void* d_out, int out_size, void* d_ws, size_t ws_size,
                              hipStream_t stream) {
    const float* x_q    = (const float*)d_in[0];
    const float* x_kv   = (const float*)d_in[1];
    const float* w_q    = (const float*)d_in[2];
    const float* w_kv   = (const float*)d_in[3];
    const float* w_qdw  = (const float*)d_in[4];
    const float* w_kvdw = (const float*)d_in[5];
    const float* w_out  = (const float*)d_in[6];
    const float* temp   = (const float*)d_in[7];
    float* out = (float*)d_out;
    float* ws  = (float*)d_ws;

    const size_t S1 = (size_t)NB * CQ * HW;   // one [B,192,HW] plane (f32 elems)
    float* q_pre    = ws;                     // stage-1 Q projection
    float* kv_pre   = ws + S1;                // stage-1 KV projection (2*S1)
    float* q_dw     = ws + 3 * S1;            // depthwise Q
    float* kv_dw    = ws + 4 * S1;            // depthwise KV (2*S1): k then v
    float* attn_out = ws;                     // aliases q_pre (dead after dwconv)
    float* invq     = ws + 6 * S1;            // 1536 used, 2048 reserved
    float* invk     = invq + 2048;
    _Float16* WqH   = (_Float16*)(invk + 2048);      // 192*192
    _Float16* WkvH  = WqH  + (size_t)2 * CQ * CQ / 2 * 2;   // = WqH + 73728
    _Float16* WoutH = WkvH + (size_t)2 * CQ * CQ;            // 384*192
    _Float16* Ph    = WoutH + (size_t)CQ * CQ;               // 48*32*32 f16

    // pre-convert weights to f16
    cvt_f16<<<(CQ * CQ) / 256, 256, 0, stream>>>(w_q, WqH, CQ * CQ);
    cvt_f16<<<(2 * CQ * CQ) / 256, 256, 0, stream>>>(w_kv, WkvH, 2 * CQ * CQ);
    cvt_f16<<<(CQ * CQ) / 256, 256, 0, stream>>>(w_out, WoutH, CQ * CQ);

    // 1x1 projections (WMMA GEMMs, TDM-staged B tiles)
    pw_gemm<12><<<dim3(HW / 64, NB), 128, 0, stream>>>(WqH,  x_q,  q_pre);
    pw_gemm<24><<<dim3(HW / 64, NB), 128, 0, stream>>>(WkvH, x_kv, kv_pre);

    // depthwise 3x3
    dwconv3x3<<<(int)((size_t)NB * CQ     * HW / 256), 256, 0, stream>>>(q_pre,  w_qdw,  q_dw,  CQ);
    dwconv3x3<<<(int)((size_t)NB * 2 * CQ * HW / 256), 256, 0, stream>>>(kv_pre, w_kvdw, kv_dw, 2 * CQ);

    // inverse L2 norms of q and k channel planes
    plane_invnorm<<<NB * CQ, 256, 0, stream>>>(q_dw,  invq, CQ);
    plane_invnorm<<<NB * CQ, 256, 0, stream>>>(kv_dw, invk, 2 * CQ);

    // scores + softmax (P emitted as f16), then apply to V
    attn_scores<<<NB * NH, 256, 0, stream>>>(q_dw, kv_dw, invq, invk, temp, Ph);
    attn_apply<<<dim3(HW / 16 / 32, NB * NH), 256, 0, stream>>>(Ph, kv_dw, attn_out);

    // output 1x1 projection
    pw_gemm<12><<<dim3(HW / 64, NB), 128, 0, stream>>>(WoutH, attn_out, out);
}
